// Block_65755949302215
// MI455X (gfx1250) — compile-verified, matrix operand order checked by reading, not run on hardware
//
#include <hip/hip_runtime.h>

typedef unsigned short u16;
typedef unsigned int   u32;
typedef unsigned long long u64;
typedef __attribute__((ext_vector_type(16))) __bf16 v16bf;
typedef __attribute__((ext_vector_type(8)))  float  v8f;
typedef __attribute__((ext_vector_type(4)))  unsigned int u32x4;

union Frag { v16bf bf; u32x4 u[2]; };

__device__ __forceinline__ u16 f2bf(float f) {
    u32 u = __builtin_bit_cast(u32, f);
    u += 0x7fffu + ((u >> 16) & 1u);
    return (u16)(u >> 16);
}

// --- CDNA5 async global->LDS (ASYNCcnt) ---
__device__ __forceinline__ unsigned lds_off(const void* p) {
    return (unsigned)(uintptr_t)p;   // LDS aperture: addr[31:0] == wave LDS byte offset
}
__device__ __forceinline__ void async_b128(unsigned lds, u64 gaddr) {
    asm volatile("global_load_async_to_lds_b128 %0, %1, off"
                 :: "v"(lds), "v"(gaddr) : "memory");
}
__device__ __forceinline__ void async_wait0() {
    asm volatile("s_wait_asynccnt 0x0" ::: "memory");
}

// ---------------- one-time weight convert+transpose: in[K][N] f32 -> out[N][K] bf16 ----------------
__global__ __launch_bounds__(256) void cvt_t_bf16_kernel(const float* __restrict__ in,
                                                         u16* __restrict__ out, int K, int N) {
    int k = blockIdx.x * 32 + (threadIdx.x & 31);
    int n = blockIdx.y * 8 + (threadIdx.x >> 5);
    out[(size_t)n * K + k] = f2bf(in[(size_t)k * N + n]);
}

// ---------------- LayerNorm (row = 768), one wave per row, bf16 out ----------------
__global__ __launch_bounds__(256) void ln_bf16_kernel(const float* __restrict__ x,
                                                      const float* __restrict__ g,
                                                      const float* __restrict__ b,
                                                      u16* __restrict__ out) {
    int row  = blockIdx.x * 8 + (threadIdx.x >> 5);
    int lane = threadIdx.x & 31;
    const float* xr = x + (size_t)row * 768;
    float v[24];
    float s = 0.f;
#pragma unroll
    for (int i = 0; i < 24; i++) { v[i] = xr[lane + i * 32]; s += v[i]; }
#pragma unroll
    for (int m = 16; m >= 1; m >>= 1) s += __shfl_xor(s, m, 32);
    float mu = s * (1.f / 768.f);
    float vs = 0.f;
#pragma unroll
    for (int i = 0; i < 24; i++) { float d = v[i] - mu; vs += d * d; }
#pragma unroll
    for (int m = 16; m >= 1; m >>= 1) vs += __shfl_xor(vs, m, 32);
    float rstd = rsqrtf(vs * (1.f / 768.f) + 1e-5f);
    u16* o = out + (size_t)row * 768;
#pragma unroll
    for (int i = 0; i < 24; i++) {
        int c = lane + i * 32;
        o[c] = f2bf((v[i] - mu) * rstd * g[c] + b[c]);
    }
}

// ---------------- bf16 GEMM, async double-buffered (K-step 64), fused epilogue ----------------
// C[M,N] = A[M,K](bf16) * BT[N,K](bf16)^T + bias
// EPI bit0: store bf16 ; bit1: gelu ; bit2: add residual(f32) ; bit3: qkv split (V -> vt[B,H,d,T])
template <int EPI>
__global__ __launch_bounds__(256) void gemm_bf16_kernel(const u16* __restrict__ A,
                                                        const u16* __restrict__ BT,
                                                        const float* __restrict__ bias,
                                                        const float* __restrict__ res,
                                                        void* __restrict__ outp,
                                                        u16* __restrict__ vt,
                                                        int M, int N, int K) {
    __shared__ u16 As[2][128 * 72];   // [m][k0..k0+63], stride 72
    __shared__ u16 Bs[2][128 * 72];   // [n][k0..k0+63], stride 72

    const int tid  = threadIdx.x;
    const int lane = tid & 31;
    const int wv   = tid >> 5;
    const int hf   = lane >> 4;
    const int ln   = lane & 15;
    const int m0   = blockIdx.y * 128;
    const int n0   = blockIdx.x * 128;
    const int wm   = wv & 1;
    const int wn   = wv >> 1;
    const int srow = tid >> 1;
    const int sseg = (tid & 1) * 32;  // 32 u16 = 64B = 4 b128 per thread per matrix

    v8f acc[4][2] = {};

    auto stage = [&](int p, int k0) {
        unsigned la = lds_off(&As[p][srow * 72 + sseg]);
        u64 ga = (u64)(uintptr_t)(A + (size_t)(m0 + srow) * K + k0 + sseg);
        async_b128(la, ga);           async_b128(la + 16, ga + 16);
        async_b128(la + 32, ga + 32); async_b128(la + 48, ga + 48);
        unsigned lb = lds_off(&Bs[p][srow * 72 + sseg]);
        u64 gb = (u64)(uintptr_t)(BT + (size_t)(n0 + srow) * K + k0 + sseg);
        async_b128(lb, gb);           async_b128(lb + 16, gb + 16);
        async_b128(lb + 32, gb + 32); async_b128(lb + 48, gb + 48);
    };

    stage(0, 0);
    int p = 0;
    for (int k0 = 0; k0 < K; k0 += 64) {
        async_wait0();
        __syncthreads();
        if (k0 + 64 < K) stage(p ^ 1, k0 + 64);

#pragma unroll
        for (int ks = 0; ks < 2; ks++) {
            Frag af[4], bfr[2];
#pragma unroll
            for (int mi = 0; mi < 4; mi++) {
                const u16* q = &As[p][(wm * 64 + mi * 16 + ln) * 72 + ks * 32 + hf * 8];
                af[mi].u[0] = *(const u32x4*)q;
                af[mi].u[1] = *(const u32x4*)(q + 16);
            }
#pragma unroll
            for (int ni = 0; ni < 2; ni++) {
                const u16* q = &Bs[p][(wn * 32 + ni * 16 + ln) * 72 + ks * 32 + hf * 8];
                bfr[ni].u[0] = *(const u32x4*)q;
                bfr[ni].u[1] = *(const u32x4*)(q + 16);
            }
#pragma unroll
            for (int mi = 0; mi < 4; mi++)
#pragma unroll
                for (int ni = 0; ni < 2; ni++)
                    acc[mi][ni] = __builtin_amdgcn_wmma_f32_16x16x32_bf16(
                        false, af[mi].bf, false, bfr[ni].bf, (short)0, acc[mi][ni], false, false);
        }
        p ^= 1;
    }

    // epilogue
#pragma unroll
    for (int mi = 0; mi < 4; mi++) {
        int mb = m0 + wm * 64 + mi * 16 + hf * 8;
#pragma unroll
        for (int ni = 0; ni < 2; ni++) {
            int n = n0 + wn * 32 + ni * 16 + ln;
            float bv = bias[n];
#pragma unroll
            for (int r = 0; r < 8; r++) {
                float v = acc[mi][ni][r] + bv;
                if (EPI & 2) {
                    v = 0.5f * v * (1.0f + tanhf(0.7978845608028654f * (v + 0.044715f * v * v * v)));
                }
                if (EPI & 8) {
                    if (n < 1536) {
                        ((u16*)outp)[(size_t)(mb + r) * N + n] = f2bf(v);
                    } else {                     // V head: store transposed [B,H,d,T]
                        int col = n - 1536, hh = col >> 6, dd = col & 63;
                        int bb = mb >> 11, t0 = mb & 2047;
                        vt[((size_t)(bb * 12 + hh) * 64 + dd) * 2048 + t0 + r] = f2bf(v);
                    }
                } else {
                    size_t idx = (size_t)(mb + r) * N + n;
                    float vv = v;
                    if (EPI & 4) vv += res[idx];
                    if (EPI & 1) ((u16*)outp)[idx] = f2bf(vv);
                    else         ((float*)outp)[idx] = vv;
                }
            }
        }
    }
}

// ---------------- flash attention, causal, hd=64, H=12, T=2048 ----------------
// qkv: [B,T,3C] bf16 (Q,K used) ; vt: [B,H,64,T] bf16 (V transposed) ; cv: [B,T,C] bf16
__global__ __launch_bounds__(256) void flash_attn_kernel(const u16* __restrict__ qkv,
                                                         const u16* __restrict__ vt,
                                                         u16* __restrict__ cv) {
    const int C = 768, C3 = 2304, T = 2048;
    const int bh = blockIdx.y;
    const int b  = bh / 12, h = bh % 12;
    const int qt = blockIdx.x;
    const int tid = threadIdx.x, lane = tid & 31, wv = tid >> 5;
    const int hf = lane >> 4, ln = lane & 15;

    __shared__ u16 Ks[2][64 * 72];   // [key][d]
    __shared__ u16 Vs[2][64 * 72];   // [d][key]  (loaded directly from vt)
    __shared__ u16 Ps[8][16 * 72];

    const int q0 = qt * 128 + wv * 16;

    Frag qf[2];
#pragma unroll
    for (int ds = 0; ds < 2; ds++) {
        const u16* qp = qkv + (size_t)(b * T + q0 + ln) * C3 + h * 64 + ds * 32 + hf * 8;
        qf[ds].u[0] = *(const u32x4*)qp;
        qf[ds].u[1] = *(const u32x4*)(qp + 16);
    }

    v8f o[4] = {};
    float mrow[8], lrow[8];
#pragma unroll
    for (int r = 0; r < 8; r++) { mrow[r] = -3.0e38f; lrow[r] = 0.f; }

    const int rr = tid >> 2, seg = (tid & 3) * 16;
    auto stageKV = [&](int p, int kt) {
        unsigned lk = lds_off(&Ks[p][rr * 72 + seg]);
        u64 gk = (u64)(uintptr_t)(qkv + (size_t)(b * T + kt * 64 + rr) * C3 + C + h * 64 + seg);
        async_b128(lk, gk);
        async_b128(lk + 16, gk + 16);
        unsigned lv = lds_off(&Vs[p][rr * 72 + seg]);
        u64 gv = (u64)(uintptr_t)(vt + ((size_t)(b * 12 + h) * 64 + rr) * 2048 + kt * 64 + seg);
        async_b128(lv, gv);
        async_b128(lv + 16, gv + 16);
    };

    const int nkt = 2 * qt + 2;
    stageKV(0, 0);
    int p = 0;
    for (int kt = 0; kt < nkt; kt++) {
        async_wait0();
        __syncthreads();
        if (kt + 1 < nkt) stageKV(p ^ 1, kt + 1);

        // S = Q * K^T
        v8f s[4] = {};
#pragma unroll
        for (int sub = 0; sub < 4; sub++) {
#pragma unroll
            for (int ds = 0; ds < 2; ds++) {
                Frag kf;
                const u16* kp = &Ks[p][(sub * 16 + ln) * 72 + ds * 32 + hf * 8];
                kf.u[0] = *(const u32x4*)kp;
                kf.u[1] = *(const u32x4*)(kp + 16);
                s[sub] = __builtin_amdgcn_wmma_f32_16x16x32_bf16(
                    false, qf[ds].bf, false, kf.bf, (short)0, s[sub], false, false);
            }
        }

        // scale + causal mask + online softmax
#pragma unroll
        for (int sub = 0; sub < 4; sub++) {
            int kg = kt * 64 + sub * 16 + ln;
#pragma unroll
            for (int r = 0; r < 8; r++) {
                int qg = q0 + hf * 8 + r;
                float v = s[sub][r] * 0.125f;
                s[sub][r] = (kg > qg) ? -3.0e38f : v;
            }
        }
#pragma unroll
        for (int r = 0; r < 8; r++) {
            float mx = fmaxf(fmaxf(s[0][r], s[1][r]), fmaxf(s[2][r], s[3][r]));
#pragma unroll
            for (int m = 8; m >= 1; m >>= 1) mx = fmaxf(mx, __shfl_xor(mx, m, 32));
            float mnew  = fmaxf(mrow[r], mx);
            float alpha = __expf(mrow[r] - mnew);
            float rsum  = 0.f;
#pragma unroll
            for (int sub = 0; sub < 4; sub++) {
                float pe = __expf(s[sub][r] - mnew);
                s[sub][r] = pe;
                rsum += pe;
            }
#pragma unroll
            for (int m = 8; m >= 1; m >>= 1) rsum += __shfl_xor(rsum, m, 32);
            mrow[r] = mnew;
            lrow[r] = lrow[r] * alpha + rsum;
#pragma unroll
            for (int dsub = 0; dsub < 4; dsub++) o[dsub][r] *= alpha;
        }

        // P (C layout) -> LDS bf16 -> reload in A layout
#pragma unroll
        for (int sub = 0; sub < 4; sub++)
#pragma unroll
            for (int r = 0; r < 8; r++)
                Ps[wv][(hf * 8 + r) * 72 + sub * 16 + ln] = f2bf(s[sub][r]);
        __syncthreads();

        // O += P * V
#pragma unroll
        for (int dsub = 0; dsub < 4; dsub++) {
#pragma unroll
            for (int ks = 0; ks < 2; ks++) {
                Frag pf, vf;
                const u16* pp = &Ps[wv][ln * 72 + ks * 32 + hf * 8];
                pf.u[0] = *(const u32x4*)pp;
                pf.u[1] = *(const u32x4*)(pp + 16);
                const u16* vp = &Vs[p][(dsub * 16 + ln) * 72 + ks * 32 + hf * 8];
                vf.u[0] = *(const u32x4*)vp;
                vf.u[1] = *(const u32x4*)(vp + 16);
                o[dsub] = __builtin_amdgcn_wmma_f32_16x16x32_bf16(
                    false, pf.bf, false, vf.bf, (short)0, o[dsub], false, false);
            }
        }
        p ^= 1;
    }

    // finalize
#pragma unroll
    for (int dsub = 0; dsub < 4; dsub++) {
#pragma unroll
        for (int r = 0; r < 8; r++) {
            float v = o[dsub][r] / lrow[r];
            int q = q0 + hf * 8 + r;
            cv[(size_t)(b * T + q) * C + h * 64 + dsub * 16 + ln] = f2bf(v);
        }
    }
}

// ---------------- host side ----------------
extern "C" void kernel_launch(void* const* d_in, const int* in_sizes, int n_in,
                              void* d_out, int out_size, void* d_ws, size_t ws_size,
                              hipStream_t stream) {
    const int C = 768, C3 = 2304, C4 = 3072, T = 2048, B = 4;
    const int R = B * T;

    const float* x      = (const float*)d_in[0];
    const float* ln1_g  = (const float*)d_in[1];
    const float* ln1_b  = (const float*)d_in[2];
    const float* w_attn = (const float*)d_in[3];
    const float* b_attn = (const float*)d_in[4];
    const float* w_proj = (const float*)d_in[5];
    const float* b_proj = (const float*)d_in[6];
    const float* ln2_g  = (const float*)d_in[7];
    const float* ln2_b  = (const float*)d_in[8];
    const float* w_fc   = (const float*)d_in[9];
    const float* b_fc   = (const float*)d_in[10];
    const float* w_fc2  = (const float*)d_in[11];
    const float* b_fc2  = (const float*)d_in[12];

    char* ws = (char*)d_ws;
    size_t off = 0;
    auto take = [&](size_t bytes) -> char* {
        char* p = ws + off;
        off += (bytes + 255) & ~(size_t)255;
        return p;
    };
    u16*   waT   = (u16*)take((size_t)C * C3 * 2);   // [2304][768]
    u16*   wpT   = (u16*)take((size_t)C * C  * 2);   // [768][768]
    u16*   wfcT  = (u16*)take((size_t)C * C4 * 2);   // [3072][768]
    u16*   wfc2T = (u16*)take((size_t)C4 * C * 2);   // [768][3072]
    u16*   h1    = (u16*)take((size_t)R * C  * 2);
    u16*   qkv   = (u16*)take((size_t)R * C3 * 2);
    u16*   vt    = (u16*)take((size_t)B * 12 * 64 * T * 2);
    u16*   cvb   = (u16*)take((size_t)R * C  * 2);
    float* x2    = (float*)take((size_t)R * C * 4);
    u16*   h2    = (u16*)take((size_t)R * C  * 2);
    u16*   hg    = (u16*)take((size_t)R * C4 * 2);

    // 1) weights -> bf16, transposed to [N][K]
    cvt_t_bf16_kernel<<<dim3(C / 32, C3 / 8), 256, 0, stream>>>(w_attn, waT, C, C3);
    cvt_t_bf16_kernel<<<dim3(C / 32, C  / 8), 256, 0, stream>>>(w_proj, wpT, C, C);
    cvt_t_bf16_kernel<<<dim3(C / 32, C4 / 8), 256, 0, stream>>>(w_fc,   wfcT, C, C4);
    cvt_t_bf16_kernel<<<dim3(C4 / 32, C / 8), 256, 0, stream>>>(w_fc2,  wfc2T, C4, C);
    // 2) h1 = LN1(x)
    ln_bf16_kernel<<<R / 8, 256, 0, stream>>>(x, ln1_g, ln1_b, h1);
    // 3) qkv = h1 @ w_attn + b_attn   (Q,K -> qkv bf16; V -> vt transposed)
    gemm_bf16_kernel<1 | 8><<<dim3(C3 / 128, R / 128), 256, 0, stream>>>(h1, waT, b_attn, nullptr, qkv, vt, R, C3, C);
    // 4) cv = causal attention
    flash_attn_kernel<<<dim3(T / 128, B * 12), 256, 0, stream>>>(qkv, vt, cvb);
    // 5) x2 = cv @ w_proj + b_proj + x
    gemm_bf16_kernel<4><<<dim3(C / 128, R / 128), 256, 0, stream>>>(cvb, wpT, b_proj, x, x2, nullptr, R, C, C);
    // 6) h2 = LN2(x2)
    ln_bf16_kernel<<<R / 8, 256, 0, stream>>>(x2, ln2_g, ln2_b, h2);
    // 7) hg = gelu(h2 @ w_fc + b_fc)
    gemm_bf16_kernel<1 | 2><<<dim3(C4 / 128, R / 128), 256, 0, stream>>>(h2, wfcT, b_fc, nullptr, hg, nullptr, R, C4, C);
    // 8) out = hg @ w_fc2 + b_fc2 + x2
    gemm_bf16_kernel<4><<<dim3(C / 128, R / 128), 256, 0, stream>>>(hg, wfc2T, b_fc2, x2, d_out, nullptr, R, C, C4);
}
